// ROI_ALIGN_9380208574628
// MI455X (gfx1250) — compile-verified, matrix operand order checked by reading, not run on hardware
//
#include <hip/hip_runtime.h>
#include <hip/hip_bf16.h>

typedef __attribute__((ext_vector_type(16))) _Float16 v16h;
typedef __attribute__((ext_vector_type(8)))  float    v8f;
typedef __attribute__((ext_vector_type(4)))  _Float16 h4;

#define POOLED 7
#define SCALE  0.25f
#define C_DIM  256
#define H_DIM  128
#define W_DIM  128
#define PATCH  16
#define WSTR   264   // W row stride (halves): mult of 8 -> 16B-aligned b128 reads, bank-spread
#define PSTR   264
#define CCHUNK 32    // channels per LDS patch chunk

union V16U { v16h h; uint4 q[2]; };

__global__ __launch_bounds__(256) void roi_align_wmma_kernel(
    const float* __restrict__ feat, const float* __restrict__ boxes,
    float* __restrict__ out)
{
    __shared__ _Float16 Wlds[64 * WSTR];       // dense weights: 64 bins x 256 patch cells
    __shared__ _Float16 Plds[CCHUNK * PSTR];   // patch chunk: 32 channels x 256 cells

    const int k    = blockIdx.x;
    const int tid  = threadIdx.x;
    const int lane = tid & 31;
    const int wv   = tid >> 5;

    // ---- uniform per-ROI parameters ----
    const float* bx = boxes + (size_t)k * 5;
    const int   bi  = (int)bx[0];
    const float x1  = bx[1] * SCALE, y1 = bx[2] * SCALE;
    const float x2  = bx[3] * SCALE, y2 = bx[4] * SCALE;
    const float roi_w = fmaxf(x2 - x1, 1.0f);
    const float roi_h = fmaxf(y2 - y1, 1.0f);
    const float bin_w = roi_w * (1.0f / POOLED);
    const float bin_h = roi_h * (1.0f / POOLED);
    const float gwf = ceilf(roi_w * (1.0f / POOLED));
    const float ghf = ceilf(roi_h * (1.0f / POOLED));
    const int   gw  = min(max((int)gwf, 1), 2);   // GMAX = 2
    const int   gh  = min(max((int)ghf, 1), 2);
    const float inv = 1.0f / (ghf * gwf);
    int x0 = (int)floorf(x1); x0 = min(max(x0, 0), W_DIM - PATCH);
    int y0 = (int)floorf(y1); y0 = min(max(y0, 0), H_DIM - PATCH);

    // ---- zero the weight matrix (incl. padding rows 49..63 read by WMMA) ----
    {
        unsigned int* wz = (unsigned int*)Wlds;
        for (int i = tid; i < (64 * WSTR) / 2; i += 256) wz[i] = 0u;
    }
    __syncthreads();

    // ---- build dense bilinear weight rows: one thread per bin ----
    if (tid < 49) {
        const int ph = tid / 7, pw = tid % 7;
        _Float16* wr = Wlds + tid * WSTR;
        for (int iy = 0; iy < gh; ++iy) {
            const float Y = y1 + ph * bin_h + (iy + 0.5f) * bin_h / ghf;
            for (int ix = 0; ix < gw; ++ix) {
                const float X = x1 + pw * bin_w + (ix + 0.5f) * bin_w / gwf;
                if (Y < -1.0f || Y > (float)H_DIM || X < -1.0f || X > (float)W_DIM)
                    continue;                           // joint sample invalidation (m = 0)
                float Yc = fmaxf(Y, 0.0f), Xc = fmaxf(X, 0.0f);
                const int yl = min((int)Yc, H_DIM - 1);
                const int xl = min((int)Xc, W_DIM - 1);
                const int yh = min(yl + 1, H_DIM - 1);
                const int xh = min(xl + 1, W_DIM - 1);
                if (yl >= H_DIM - 1) Yc = (float)yl;    // torchvision border semantics
                if (xl >= W_DIM - 1) Xc = (float)xl;
                const float ly = Yc - (float)yl, lx = Xc - (float)xl;
                const float hy = 1.0f - ly,      hx = 1.0f - lx;
                const int ry0 = yl - y0, rx0 = xl - x0;
                const int ry1 = yh - y0, rx1 = xh - x0;
                const float w1 = hy * hx * inv, w2 = hy * lx * inv;
                const float w3 = ly * hx * inv, w4 = ly * lx * inv;
                if ((unsigned)ry0 < 16u && (unsigned)rx0 < 16u) {
                    const int s = ry0 * 16 + rx0; wr[s] = (_Float16)((float)wr[s] + w1);
                }
                if ((unsigned)ry0 < 16u && (unsigned)rx1 < 16u) {
                    const int s = ry0 * 16 + rx1; wr[s] = (_Float16)((float)wr[s] + w2);
                }
                if ((unsigned)ry1 < 16u && (unsigned)rx0 < 16u) {
                    const int s = ry1 * 16 + rx0; wr[s] = (_Float16)((float)wr[s] + w3);
                }
                if ((unsigned)ry1 < 16u && (unsigned)rx1 < 16u) {
                    const int s = ry1 * 16 + rx1; wr[s] = (_Float16)((float)wr[s] + w4);
                }
            }
        }
    }

    // ---- WMMA tiling: wave -> (mtile over 64 bins, ctile over 32-chan chunk) ----
    const int mtile = wv >> 1;            // 0..3 : bins [16*mtile, 16*mtile+15]
    const int ctile = wv & 1;             // 0..1 : channels within chunk
    const int nrow  = lane & 15;          // A row (M) and B column (N)
    const int khi   = (lane & 16) ? 8 : 0;  // 16B K-chunk lane-half offset (f16 layout)
    const int mhi   = (lane & 16) ? 8 : 0;  // D-matrix M offset for lanes 16..31

    for (int cbase = 0; cbase < C_DIM; cbase += CCHUNK) {
        // ---- gather 16x16 patch for CCHUNK channels into LDS (f32 -> f16) ----
        for (int it = 0; it < (CCHUNK * 16 * 4) / 256; ++it) {
            const int q   = it * 256 + tid;       // quad id: chunk has 2048 float4-quads
            const int cl  = q >> 6;               // channel within chunk
            const int rem = q & 63;
            const int dy  = rem >> 2;
            const int qx  = (rem & 3) << 2;
            const float* src = feat +
                (((size_t)(bi * C_DIM + cbase + cl) * H_DIM) + (size_t)(y0 + dy)) * W_DIM
                + x0 + qx;
            h4 hv = {(_Float16)src[0], (_Float16)src[1], (_Float16)src[2], (_Float16)src[3]};
            *(h4*)(Plds + cl * PSTR + dy * 16 + qx) = hv;   // 8B packed ds store
            if (cbase + CCHUNK < C_DIM)                      // warm next chunk in L2
                __builtin_prefetch(src + (size_t)CCHUNK * H_DIM * W_DIM, 0, 1);
        }
        __syncthreads();

        // ---- D[bin, ch] += A(W) x B(P) over K = 256 patch cells, 8 chunks of 32 ----
        v8f acc = {0.f, 0.f, 0.f, 0.f, 0.f, 0.f, 0.f, 0.f};
        const _Float16* wrow = Wlds + (mtile * 16 + nrow) * WSTR;
        const _Float16* prow = Plds + (ctile * 16 + nrow) * PSTR;
        #pragma unroll
        for (int kc = 0; kc < 8; ++kc) {
            const int kb = kc * 32 + khi;
            V16U a, b;
            a.q[0] = *(const uint4*)(wrow + kb);        // K = kb   .. kb+7
            a.q[1] = *(const uint4*)(wrow + kb + 16);   // K = kb+16.. kb+23
            b.q[0] = *(const uint4*)(prow + kb);
            b.q[1] = *(const uint4*)(prow + kb + 16);
            acc = __builtin_amdgcn_wmma_f32_16x16x32_f16(
                false, a.h, false, b.h, (short)0, acc, false, false);
        }

        // ---- store: D vgpr r -> bin = mtile*16 + mhi + r, out[k][ch][ph][pw] ----
        const int ch = cbase + ctile * 16 + nrow;
        float* orow = out + ((size_t)k * C_DIM + ch) * 49;
        #pragma unroll
        for (int r = 0; r < 8; ++r) {
            const int bin = mtile * 16 + mhi + r;
            if (bin < 49) orow[bin] = acc[r];
        }
        __syncthreads();   // P reused next chunk
    }
}

extern "C" void kernel_launch(void* const* d_in, const int* in_sizes, int n_in,
                              void* d_out, int out_size, void* d_ws, size_t ws_size,
                              hipStream_t stream) {
    const float* feat  = (const float*)d_in[0];
    const float* boxes = (const float*)d_in[1];
    float* out = (float*)d_out;
    const int K = in_sizes[1] / 5;
    roi_align_wmma_kernel<<<K, 256, 0, stream>>>(feat, boxes, out);
}
